// MultiheadAttentionLayer_20572893347962
// MI455X (gfx1250) — matrix operand order repaired and checked by reading, not compiled, not run on hardware
//
#include <hip/hip_runtime.h>
#include <hip/hip_bf16.h>

typedef __bf16 bf16_t;
typedef __attribute__((ext_vector_type(16))) __bf16 v16bf;
typedef __attribute__((ext_vector_type(8)))  __bf16 v8bf;
typedef __attribute__((ext_vector_type(8)))  float  v8f;

#define HID   1024
#define NH    16
#define HD    64
#define SEQ   2048
#define BATCH 2
#define MROWS (BATCH*SEQ)   // 4096

static __device__ inline v8f wmma_bf16(v16bf a, v16bf b, v8f c) {
  // D = A(16x32 bf16) * B(32x16 bf16) + C(16x16 f32)
  return __builtin_amdgcn_wmma_f32_16x16x32_bf16(false, a, false, b, (short)0, c,
                                                 false, false);
}

// A-fragment (16x32, 16-bit): lane L: m=L%16; element e -> k = (L/16)*8 + (e&7) + (e>=8)*16
// caller passes kbase = kchunk + (L/16)*8 ; loads runs [kbase,kbase+8) and [kbase+16,kbase+24)
static __device__ inline v16bf load_a_f32(const float* __restrict__ row, int kbase) {
  float t[16];
  *(float4*)(t + 0)  = *(const float4*)(row + kbase + 0);
  *(float4*)(t + 4)  = *(const float4*)(row + kbase + 4);
  *(float4*)(t + 8)  = *(const float4*)(row + kbase + 16);
  *(float4*)(t + 12) = *(const float4*)(row + kbase + 20);
  v16bf a;
  #pragma unroll
  for (int e = 0; e < 16; ++e) a[e] = (bf16_t)t[e];
  return a;
}

// B-fragment (32x16, 16-bit): lane L: n=L%16; element e -> k = (L/16)*8 + (e&7) + (e>=8)*16
// caller passes kbase = kchunk + (L/16)*8
static __device__ inline v16bf load_b_f32(const float* __restrict__ row, int kbase) {
  float t[16];
  #pragma unroll
  for (int i = 0; i < 4; ++i)
    *(float4*)(t + 4*i) = *(const float4*)(row + kbase + 4*i + (i >= 2) * 8);
  v16bf b;
  #pragma unroll
  for (int e = 0; e < 16; ++e) b[e] = (bf16_t)t[e];
  return b;
}

static __device__ inline v16bf load_a_bf(const bf16_t* __restrict__ row, int kbase) {
  v8bf lo = *(const v8bf*)(row + kbase);
  v8bf hi = *(const v8bf*)(row + kbase + 16);
  v16bf a;
  #pragma unroll
  for (int e = 0; e < 8; ++e) { a[e] = lo[e]; a[e + 8] = hi[e]; }
  return a;
}

static __device__ inline v16bf load_b_bf(const bf16_t* __restrict__ row, int kbase) {
  v8bf lo = *(const v8bf*)(row + kbase);
  v8bf hi = *(const v8bf*)(row + kbase + 16);
  v16bf b;
  #pragma unroll
  for (int e = 0; e < 8; ++e) { b[e] = lo[e]; b[e + 8] = hi[e]; }
  return b;
}

// ---------------------------------------------------------------------------
// out[m,n] = sum_k A[m,k] * W[n,k] + bias[n]      (torch Linear: x @ W.T + b)
// wave tile: 32 (M) x 64 (N); 8 waves / block.
// ---------------------------------------------------------------------------
__global__ __launch_bounds__(256)
void gemm_xwt_kernel(const float* __restrict__ A, const float* __restrict__ W,
                     const float* __restrict__ bias,
                     float* __restrict__ outF, bf16_t* __restrict__ outB,
                     bf16_t* __restrict__ outT,  // [N][M] transposed bf16 (optional)
                     int M, int N, int K) {
  const int lane = threadIdx.x & 31;
  const int half = lane >> 4;
  const int l16  = lane & 15;
  const int gw   = blockIdx.x * (blockDim.x >> 5) + (threadIdx.x >> 5);
  const int mtiles = M >> 5;
  const int mi = gw % mtiles, ni = gw / mtiles;
  if (ni >= (N >> 6)) return;
  const int mbase = mi * 32, nbase = ni * 64;

  const float* arow0 = A + (size_t)(mbase + l16) * K;
  const float* arow1 = arow0 + (size_t)16 * K;
  const float* wr[4];
  #pragma unroll
  for (int j = 0; j < 4; ++j) wr[j] = W + (size_t)(nbase + j * 16 + l16) * K;

  v8f acc[2][4];
  #pragma unroll
  for (int p = 0; p < 2; ++p)
    #pragma unroll
    for (int j = 0; j < 4; ++j)
      acc[p][j] = (v8f){0.f,0.f,0.f,0.f,0.f,0.f,0.f,0.f};

  for (int k0 = 0; k0 < K; k0 += 32) {
    v16bf a0 = load_a_f32(arow0, k0 + half * 8);
    v16bf a1 = load_a_f32(arow1, k0 + half * 8);
    #pragma unroll
    for (int j = 0; j < 4; ++j) {
      v16bf bj = load_b_f32(wr[j], k0 + half * 8);
      acc[0][j] = wmma_bf16(a0, bj, acc[0][j]);
      acc[1][j] = wmma_bf16(a1, bj, acc[1][j]);
    }
  }

  // C/D layout: VGPR r -> m = r + 8*(lane/16); n = lane%16
  #pragma unroll
  for (int p = 0; p < 2; ++p) {
    #pragma unroll
    for (int j = 0; j < 4; ++j) {
      const int n = nbase + j * 16 + l16;
      const float bv = bias[n];
      #pragma unroll
      for (int r = 0; r < 8; ++r) {
        const int m = mbase + p * 16 + half * 8 + r;
        const float v = acc[p][j][r] + bv;
        if (outF) outF[(size_t)m * N + n] = v;
        if (outB) outB[(size_t)m * N + n] = (bf16_t)v;
        if (outT) outT[(size_t)n * M + m] = (bf16_t)v;
      }
    }
  }
}

// ---------------------------------------------------------------------------
// One block per (batch b, head h, 16 query rows). 8 waves split 2048 keys.
// Writes attention probs to attn_out [B,H,S,S] and heads-concat x to xcat.
// ---------------------------------------------------------------------------
__global__ __launch_bounds__(256)
void attn_kernel(const bf16_t* __restrict__ Q, const bf16_t* __restrict__ Kmat,
                 const bf16_t* __restrict__ Vt, const int* __restrict__ mask,
                 float* __restrict__ attn_out, float* __restrict__ xcat) {
  const int tid  = threadIdx.x;
  const int wid  = tid >> 5;
  const int lane = tid & 31;
  const int half = lane >> 4;
  const int l16  = lane & 15;

  const int qblocks = SEQ / 16;
  const int qt = blockIdx.x % qblocks;
  const int h  = (blockIdx.x / qblocks) % NH;
  const int b  = blockIdx.x / (qblocks * NH);

  __shared__ float red[8][16];
  __shared__ float xred[16][HD];
  __shared__ __align__(16) bf16_t patch[8][16 * 32];

  // Q A-fragments: rows m = l16, d chunks [0,32) and [32,64)
  const bf16_t* qrow = Q + ((size_t)(b * SEQ + qt * 16 + l16) * HID + h * HD);
  const v16bf a0 = load_a_bf(qrow, half * 8);
  const v16bf a1 = load_a_bf(qrow, 32 + half * 8);

  const int kb0 = wid * 256;          // this wave's 256 keys
  v8f energy[16];

  // ---- energy = (Q K^T)/8, masked -------------------------------------
  #pragma unroll
  for (int t = 0; t < 16; ++t) {
    const int kb = kb0 + t * 16;
    const bf16_t* krow = Kmat + ((size_t)(b * SEQ + kb + l16) * HID + h * HD);
    v16bf kb_0 = load_b_bf(krow, half * 8);         // d 0..31
    v16bf kb_1 = load_b_bf(krow, 32 + half * 8);    // d 32..63
    v8f e = (v8f){0.f,0.f,0.f,0.f,0.f,0.f,0.f,0.f};
    e = wmma_bf16(a0, kb_0, e);
    e = wmma_bf16(a1, kb_1, e);
    const size_t mb = (size_t)b * SEQ * SEQ +
                      (size_t)(qt * 16 + half * 8) * SEQ + kb + l16;
    #pragma unroll
    for (int r = 0; r < 8; ++r) {
      const int mv = mask[mb + (size_t)r * SEQ];
      const float ev = e[r] * 0.125f;               // / sqrt(64)
      e[r] = (mv == 0) ? -1e10f : ev;
    }
    energy[t] = e;
  }

  // ---- softmax: row max (half-wave shuffle, then cross-wave via LDS) --
  float rowmax[8];
  #pragma unroll
  for (int r = 0; r < 8; ++r) {
    float m = energy[0][r];
    #pragma unroll
    for (int t = 1; t < 16; ++t) m = fmaxf(m, energy[t][r]);
    m = fmaxf(m, __shfl_xor(m, 1, 32));
    m = fmaxf(m, __shfl_xor(m, 2, 32));
    m = fmaxf(m, __shfl_xor(m, 4, 32));
    m = fmaxf(m, __shfl_xor(m, 8, 32));
    rowmax[r] = m;
  }
  if (l16 == 0) {
    #pragma unroll
    for (int r = 0; r < 8; ++r) red[wid][half * 8 + r] = rowmax[r];
  }
  __syncthreads();
  #pragma unroll
  for (int r = 0; r < 8; ++r) {
    float m = red[0][half * 8 + r];
    #pragma unroll
    for (int w = 1; w < 8; ++w) m = fmaxf(m, red[w][half * 8 + r]);
    rowmax[r] = m;
  }
  __syncthreads();  // reads of red done before reuse for sums

  // ---- exp + row sum ---------------------------------------------------
  float rsum[8];
  #pragma unroll
  for (int r = 0; r < 8; ++r) rsum[r] = 0.f;
  #pragma unroll
  for (int t = 0; t < 16; ++t)
    #pragma unroll
    for (int r = 0; r < 8; ++r) {
      const float p = __expf(energy[t][r] - rowmax[r]);
      energy[t][r] = p;
      rsum[r] += p;
    }
  #pragma unroll
  for (int r = 0; r < 8; ++r) {
    float s = rsum[r];
    s += __shfl_xor(s, 1, 32);
    s += __shfl_xor(s, 2, 32);
    s += __shfl_xor(s, 4, 32);
    s += __shfl_xor(s, 8, 32);
    rsum[r] = s;
  }
  if (l16 == 0) {
    #pragma unroll
    for (int r = 0; r < 8; ++r) red[wid][half * 8 + r] = rsum[r];
  }
  __syncthreads();
  float rinv[8];
  #pragma unroll
  for (int r = 0; r < 8; ++r) {
    float s = red[0][half * 8 + r];
    #pragma unroll
    for (int w = 1; w < 8; ++w) s += red[w][half * 8 + r];
    rinv[r] = 1.0f / s;
  }

  // zero x reduction buffer (barrier before atomics comes after phase 4)
  for (int i = tid; i < 16 * HD; i += 256) ((float*)xred)[i] = 0.f;

  // ---- write attention, compute x += P V (32 keys per step) -----------
  v8f xacc[4];
  #pragma unroll
  for (int j = 0; j < 4; ++j) xacc[j] = (v8f){0.f,0.f,0.f,0.f,0.f,0.f,0.f,0.f};

  const size_t arow = ((size_t)(b * NH + h) * SEQ + qt * 16 + half * 8) * SEQ;
  #pragma unroll
  for (int t = 0; t < 16; t += 2) {
    #pragma unroll
    for (int u = 0; u < 2; ++u) {
      const int tt = t + u;
      const int kb = kb0 + tt * 16;
      #pragma unroll
      for (int r = 0; r < 8; ++r) {
        const float att = energy[tt][r] * rinv[r];
        attn_out[arow + (size_t)r * SEQ + kb + l16] = att;
        patch[wid][(half * 8 + r) * 32 + u * 16 + l16] = (bf16_t)att;   // C->A relayout
      }
    }
    const v16bf pa = load_a_bf(&patch[wid][l16 * 32], half * 8);
    #pragma unroll
    for (int j = 0; j < 4; ++j) {
      // Vt layout [h*HD+d][b*SEQ+key]: contiguous keys per lane
      const bf16_t* vrow = Vt + ((size_t)(h * HD + j * 16 + l16) * MROWS +
                                 b * SEQ + kb0 + t * 16 + half * 8);
      const v16bf vb = load_b_bf(vrow, 0);
      xacc[j] = wmma_bf16(pa, vb, xacc[j]);
    }
  }

  __syncthreads();
  #pragma unroll
  for (int j = 0; j < 4; ++j)
    #pragma unroll
    for (int r = 0; r < 8; ++r)
      atomicAdd(&xred[half * 8 + r][j * 16 + l16], xacc[j][r]);
  __syncthreads();

  for (int i = tid; i < 16 * HD; i += 256) {
    const int m = i >> 6, d = i & 63;
    xcat[(size_t)(b * SEQ + qt * 16 + m) * HID + h * HD + d] = xred[m][d];
  }
}

extern "C" void kernel_launch(void* const* d_in, const int* in_sizes, int n_in,
                              void* d_out, int out_size, void* d_ws, size_t ws_size,
                              hipStream_t stream) {
  (void)in_sizes; (void)n_in; (void)out_size; (void)ws_size;
  const float* query = (const float*)d_in[0];
  const float* key   = (const float*)d_in[1];
  const float* value = (const float*)d_in[2];
  const int*   mask  = (const int*)d_in[3];
  const float* Wq = (const float*)d_in[4];  const float* bq = (const float*)d_in[5];
  const float* Wk = (const float*)d_in[6];  const float* bk = (const float*)d_in[7];
  const float* Wv = (const float*)d_in[8];  const float* bv = (const float*)d_in[9];
  const float* Wo = (const float*)d_in[10]; const float* bo = (const float*)d_in[11];

  bf16_t* Qw  = (bf16_t*)d_ws;                        // [4096,1024] bf16
  bf16_t* Kw  = Qw  + (size_t)MROWS * HID;            // [4096,1024] bf16
  bf16_t* VtW = Kw  + (size_t)MROWS * HID;            // [1024,4096] bf16 (transposed V)
  float*  xcat = (float*)(VtW + (size_t)MROWS * HID); // [4096,1024] f32

  float* xout = (float*)d_out;                        // [2,2048,1024]
  float* attn = xout + (size_t)MROWS * HID;           // [2,16,2048,2048]

  const int gemmBlocks = (MROWS / 32) * (HID / 64) / 8;  // 256

  gemm_xwt_kernel<<<gemmBlocks, 256, 0, stream>>>(query, Wq, bq, nullptr, Qw, nullptr,
                                                  MROWS, HID, HID);
  gemm_xwt_kernel<<<gemmBlocks, 256, 0, stream>>>(key,   Wk, bk, nullptr, Kw, nullptr,
                                                  MROWS, HID, HID);
  gemm_xwt_kernel<<<gemmBlocks, 256, 0, stream>>>(value, Wv, bv, nullptr, nullptr, VtW,
                                                  MROWS, HID, HID);
  attn_kernel<<<BATCH * NH * (SEQ / 16), 256, 0, stream>>>(Qw, Kw, VtW, mask, attn, xcat);
  gemm_xwt_kernel<<<gemmBlocks, 256, 0, stream>>>(xcat, Wo, bo, xout, nullptr, nullptr,
                                                  MROWS, HID, HID);
}
